// GRUModel_50500225466869
// MI455X (gfx1250) — matrix operand order repaired and checked by reading, not compiled, not run on hardware
//
#include <hip/hip_runtime.h>
#include <hip/hip_bf16.h>

// ---------------- problem dims ----------------
#define BATCH 256
#define SEQ   512
#define INDIM 512
#define HID   1024
#define OUTD  512

#define KC_X   16        // K-chunks of 32 covering INDIM (x projection)
#define KC_H   32        // K-chunks of 32 covering HID   (h projection)
#define KC_TOT (KC_X + KC_H)
#define NT_H   64        // 16-wide N tiles across HID
#define NT_O   32        // 16-wide N tiles across OUTD
#define FRAG_ELEMS 512   // 32x16 bf16 elements per B fragment

// GRU persistent grid: 4 row-blocks (64 rows) x 32 col-blocks (32 cols)
#define GRU_ROWBLKS 4
#define GRU_COLBLKS 32
#define GRU_BLOCKS  (GRU_ROWBLKS * GRU_COLBLKS)   // 128
#define NT_BLK      2                             // 16-wide n-tiles per block
// per-block LDS weight stash: 48 kc * 3 gates * 2 ntiles * 1KB = 288 KB
#define LDS_FRAGS   (KC_TOT * 3 * NT_BLK)         // 288
#define LDS_BYTES   (LDS_FRAGS * FRAG_ELEMS * 2)  // 294912

#define FC_BLOCKS  32    // 4 row-blocks x 8 col-blocks

typedef __attribute__((ext_vector_type(16))) __bf16 v16bf;
typedef __attribute__((ext_vector_type(8)))  __bf16 v8bf;
typedef __attribute__((ext_vector_type(8)))  float  v8f;
typedef int v4i __attribute__((vector_size(16)));
typedef __attribute__((address_space(1))) v4i* gvec_ptr;
typedef __attribute__((address_space(3))) v4i* lvec_ptr;

union AFrag { v16bf v; v8bf h[2]; };

// ---------------- math helpers ----------------
__device__ __forceinline__ float sigmoid_fast(float x) {
    return 1.0f / (1.0f + __expf(-x));
}
__device__ __forceinline__ float tanh_fast(float x) {
    x = fminf(fmaxf(x, -15.0f), 15.0f);
    float e = __expf(2.0f * x);
    return (e - 1.0f) / (e + 1.0f);
}

// ---------------- grid-wide barrier (persistent kernel) ----------------
__device__ __forceinline__ void grid_sync(unsigned* cnt, unsigned* gen, unsigned nblk) {
    __syncthreads();
    if (threadIdx.x == 0) {
        __threadfence();
        unsigned g = __hip_atomic_load(gen, __ATOMIC_RELAXED, __HIP_MEMORY_SCOPE_AGENT);
        unsigned a = __hip_atomic_fetch_add(cnt, 1u, __ATOMIC_ACQ_REL, __HIP_MEMORY_SCOPE_AGENT);
        if (a == nblk - 1u) {
            __hip_atomic_store(cnt, 0u, __ATOMIC_RELAXED, __HIP_MEMORY_SCOPE_AGENT);
            __hip_atomic_store(gen, g + 1u, __ATOMIC_RELEASE, __HIP_MEMORY_SCOPE_AGENT);
        } else {
            while (__hip_atomic_load(gen, __ATOMIC_ACQUIRE, __HIP_MEMORY_SCOPE_AGENT) == g) {
                __builtin_amdgcn_s_sleep(1);
            }
        }
        __threadfence();
    }
    __syncthreads();
}

// ---------------- init: zero h buffers + barrier ----------------
__global__ void gru_init_kernel(float* __restrict__ hF,
                                __bf16* __restrict__ hB0,
                                __bf16* __restrict__ hB1,
                                unsigned* __restrict__ barrier) {
    size_t i = (size_t)blockIdx.x * blockDim.x + threadIdx.x;
    size_t n = (size_t)BATCH * HID;
    if (i < n) {
        hF[i]  = 0.0f;
        hB0[i] = (__bf16)0.0f;
        hB1[i] = (__bf16)0.0f;
    }
    if (i == 0) { barrier[0] = 0u; barrier[1] = 0u; }
}

// ---------------- x: f32 -> bf16 (layout preserved [B,S,IN]) ----------------
__global__ void cvt_x_kernel(const float* __restrict__ x, __bf16* __restrict__ xb) {
    size_t n = (size_t)BATCH * SEQ * INDIM;
    for (size_t i = (size_t)blockIdx.x * blockDim.x + threadIdx.x; i < n;
         i += (size_t)gridDim.x * blockDim.x) {
        xb[i] = (__bf16)x[i];
    }
}

// B-fragment lane layout for 32x16 bf16 (wave32):
//   lane = n_local + 16*(k>>4), packed slot = k&15  ->  elem = lane*16 + slot
// Pack gates: fragment index = (kc*3 + g)*NT_H + ntile
//   kc <  KC_X : source = x-projection weight of gate g (z, r, n)
//   kc >= KC_X : source = h-projection weight of gate g
__global__ void pack_gate_weights_kernel(const float* __restrict__ xzw,
                                         const float* __restrict__ xrw,
                                         const float* __restrict__ xhw,
                                         const float* __restrict__ hzw,
                                         const float* __restrict__ hrw,
                                         const float* __restrict__ hhw,
                                         __bf16* __restrict__ wp) {
    size_t idx = (size_t)blockIdx.x * blockDim.x + threadIdx.x;
    size_t total = (size_t)KC_TOT * 3 * NT_H * FRAG_ELEMS;
    if (idx >= total) return;
    int    e    = (int)(idx & (FRAG_ELEMS - 1));
    size_t frag = idx >> 9;
    int    nt   = (int)(frag % NT_H);
    size_t gkc  = frag / NT_H;
    int    g    = (int)(gkc % 3);
    int    kc   = (int)(gkc / 3);
    int lane = e >> 4, slot = e & 15;
    int nl   = lane & 15;
    int k    = slot + ((lane >> 4) << 4);
    int col  = nt * 16 + nl;
    const float* W;
    int krow;
    if (kc < KC_X) {
        W = (g == 0) ? xzw : (g == 1) ? xrw : xhw;
        krow = kc * 32 + k;
    } else {
        W = (g == 0) ? hzw : (g == 1) ? hrw : hhw;
        krow = (kc - KC_X) * 32 + k;
    }
    wp[idx] = (__bf16)W[(size_t)krow * HID + col];
}

// Pack fc_w [HID, OUTD]: fragment index = kc*NT_O + ntile
__global__ void pack_fc_weights_kernel(const float* __restrict__ fcw,
                                       __bf16* __restrict__ wfc) {
    size_t idx = (size_t)blockIdx.x * blockDim.x + threadIdx.x;
    size_t total = (size_t)KC_H * NT_O * FRAG_ELEMS;
    if (idx >= total) return;
    int    e    = (int)(idx & (FRAG_ELEMS - 1));
    size_t frag = idx >> 9;
    int    nt   = (int)(frag % NT_O);
    int    kc   = (int)(frag / NT_O);
    int lane = e >> 4, slot = e & 15;
    int nl   = lane & 15;
    int k    = slot + ((lane >> 4) << 4);
    int col  = nt * 16 + nl;
    int krow = kc * 32 + k;
    wfc[idx] = (__bf16)fcw[(size_t)krow * OUTD + col];
}

// ---------------- persistent GRU recurrence ----------------
// Grid: 128 blocks x 128 threads (4 waves). Block tile: 64 rows x 32 cols.
// All B fragments for this block (288 KB) are staged into LDS once (via
// GLOBAL_LOAD_ASYNC_TO_LDS) and re-used across all 512 timesteps
// (MI455X: 320 KB LDS per WGP).
__global__ __launch_bounds__(128)
void gru_persistent_kernel(const __bf16* __restrict__ xbf,
                           const __bf16* __restrict__ wp,
                           const float* __restrict__ bxz, const float* __restrict__ bhz,
                           const float* __restrict__ bxr, const float* __restrict__ bhr,
                           const float* __restrict__ bxh, const float* __restrict__ bhh,
                           float* __restrict__ hF,
                           __bf16* __restrict__ hB0,
                           __bf16* __restrict__ hB1,
                           unsigned* __restrict__ barrier) {
    extern __shared__ __bf16 wlds[];   // LDS_FRAGS * FRAG_ELEMS bf16 = 288 KB

    const int lane    = threadIdx.x & 31;
    const int wave    = threadIdx.x >> 5;
    const int rowBlk  = blockIdx.x & (GRU_ROWBLKS - 1);   // 4 row blocks of 64
    const int colBlk  = blockIdx.x >> 2;                  // 32 col blocks of 32
    const int row0    = rowBlk * 64 + wave * 16;
    const int hi      = lane >> 4;                        // A/C upper-half-lane selector
    const int a_row   = row0 + (lane & 15);
    const int colBase = colBlk * 32;
    unsigned* cnt = barrier;
    unsigned* gen = barrier + 1;
    __bf16* hB[2] = { hB0, hB1 };

    // ---- prologue: stage this block's weight slice into LDS ----
    // frag_local = (kc*3+g)*NT_BLK + jn ; chunks of 16 B (8 bf16)
    {
        const int nChunks = LDS_FRAGS * (FRAG_ELEMS / 8);   // 288 * 64 = 18432
        for (int c = threadIdx.x; c < nChunks; c += blockDim.x) {
            const int fragLocal = c >> 6;          // /64 chunks per frag
            const int chunk     = c & 63;
            const int fs        = fragLocal >> 1;  // kc*3+g
            const int jn        = fragLocal & 1;
            const size_t gOff = ((size_t)fs * NT_H + (colBlk * NT_BLK + jn)) * FRAG_ELEMS
                              + (size_t)chunk * 8;
            const size_t lOff = (size_t)fragLocal * FRAG_ELEMS + (size_t)chunk * 8;
#if defined(__gfx1250__) && __has_builtin(__builtin_amdgcn_global_load_async_to_lds_b128)
            __builtin_amdgcn_global_load_async_to_lds_b128(
                (gvec_ptr)(wp + gOff),
                (lvec_ptr)(wlds + lOff),
                0, 0);
#else
            *(v8bf*)(wlds + lOff) = *(const v8bf*)(wp + gOff);
#endif
        }
#if defined(__gfx1250__) && __has_builtin(__builtin_amdgcn_global_load_async_to_lds_b128)
#if __has_builtin(__builtin_amdgcn_s_wait_asynccnt)
        __builtin_amdgcn_s_wait_asynccnt(0);
#else
        asm volatile("s_wait_asynccnt 0x0" ::: "memory");
#endif
#endif
        __syncthreads();
    }

    const v8f vzero = {0.f, 0.f, 0.f, 0.f, 0.f, 0.f, 0.f, 0.f};

    for (int t = 0; t < SEQ; ++t) {
        const __bf16* __restrict__ hread  = hB[t & 1];
        __bf16* __restrict__       hwrite = hB[(t + 1) & 1];

        v8f acc[4][NT_BLK];
        #pragma unroll
        for (int g = 0; g < 4; ++g)
            #pragma unroll
            for (int j = 0; j < NT_BLK; ++j)
                acc[g][j] = vzero;

        const __bf16* __restrict__ xrow = xbf + ((size_t)a_row * SEQ + t) * INDIM;
        const __bf16* __restrict__ hrow = hread + (size_t)a_row * HID;
        // prefetch next timestep's x slice
        __builtin_prefetch(xrow + INDIM, 0, 1);

        // ---- x-projection part of K (gates z, r, n_x), B from LDS ----
        // Hoist all 6 B fragments before the WMMA chain so DS loads overlap.
        #pragma unroll 2
        for (int kc = 0; kc < KC_X; ++kc) {
            AFrag a;
            const __bf16* p = xrow + kc * 32 + hi * 16;
            a.h[0] = *(const v8bf*)(p);
            a.h[1] = *(const v8bf*)(p + 8);
            const __bf16* lbase = wlds + (size_t)(kc * 3) * NT_BLK * FRAG_ELEMS + lane * 16;
            v16bf b[3][NT_BLK];
            #pragma unroll
            for (int g = 0; g < 3; ++g)
                #pragma unroll
                for (int j = 0; j < NT_BLK; ++j)
                    b[g][j] = *(const v16bf*)(lbase + (size_t)(g * NT_BLK + j) * FRAG_ELEMS);
            #pragma unroll
            for (int j = 0; j < NT_BLK; ++j) {
                acc[0][j] = __builtin_amdgcn_wmma_f32_16x16x32_bf16(false, a.v, false, b[0][j], (short)0, acc[0][j], false, false);
                acc[1][j] = __builtin_amdgcn_wmma_f32_16x16x32_bf16(false, a.v, false, b[1][j], (short)0, acc[1][j], false, false);
                acc[2][j] = __builtin_amdgcn_wmma_f32_16x16x32_bf16(false, a.v, false, b[2][j], (short)0, acc[2][j], false, false);
            }
        }

        // ---- h-projection part of K (gates z, r, n_h), B from LDS ----
        #pragma unroll 2
        for (int kc = 0; kc < KC_H; ++kc) {
            AFrag a;
            const __bf16* p = hrow + kc * 32 + hi * 16;
            a.h[0] = *(const v8bf*)(p);
            a.h[1] = *(const v8bf*)(p + 8);
            const __bf16* lbase = wlds + (size_t)((kc + KC_X) * 3) * NT_BLK * FRAG_ELEMS + lane * 16;
            v16bf b[3][NT_BLK];
            #pragma unroll
            for (int g = 0; g < 3; ++g)
                #pragma unroll
                for (int j = 0; j < NT_BLK; ++j)
                    b[g][j] = *(const v16bf*)(lbase + (size_t)(g * NT_BLK + j) * FRAG_ELEMS);
            #pragma unroll
            for (int j = 0; j < NT_BLK; ++j) {
                acc[0][j] = __builtin_amdgcn_wmma_f32_16x16x32_bf16(false, a.v, false, b[0][j], (short)0, acc[0][j], false, false);
                acc[1][j] = __builtin_amdgcn_wmma_f32_16x16x32_bf16(false, a.v, false, b[1][j], (short)0, acc[1][j], false, false);
                acc[3][j] = __builtin_amdgcn_wmma_f32_16x16x32_bf16(false, a.v, false, b[2][j], (short)0, acc[3][j], false, false);
            }
        }

        // ---- GRU elementwise update on owned tile ----
        #pragma unroll
        for (int j = 0; j < NT_BLK; ++j) {
            const int col = colBase + j * 16 + (lane & 15);
            const float bz  = bxz[col] + bhz[col];
            const float br  = bxr[col] + bhr[col];
            const float bnx = bxh[col];
            const float bnh = bhh[col];
            #pragma unroll
            for (int v = 0; v < 8; ++v) {
                const int row = row0 + hi * 8 + v;
                const size_t hidx = (size_t)row * HID + col;
                float z  = sigmoid_fast(acc[0][j][v] + bz);
                float r  = sigmoid_fast(acc[1][j][v] + br);
                float n  = tanh_fast(acc[2][j][v] + bnx + r * (acc[3][j][v] + bnh));
                float ho = hF[hidx];
                float hn = z * ho + (1.0f - z) * n;
                hF[hidx]     = hn;
                hwrite[hidx] = (__bf16)hn;
            }
        }

        grid_sync(cnt, gen, GRU_BLOCKS);
    }
}

// ---------------- final FC: out = h_final @ fc_w + fc_b ----------------
__global__ __launch_bounds__(128)
void fc_kernel(const __bf16* __restrict__ hB,
               const __bf16* __restrict__ wfc,
               const float* __restrict__ fcb,
               float* __restrict__ out) {
    const int lane    = threadIdx.x & 31;
    const int wave    = threadIdx.x >> 5;
    const int rowBlk  = blockIdx.x & 3;   // 4 row blocks of 64
    const int colBlk  = blockIdx.x >> 2;  // 8 col blocks of 64
    const int row0    = rowBlk * 64 + wave * 16;
    const int hi      = lane >> 4;
    const int a_row   = row0 + (lane & 15);
    const int colBase = colBlk * 64;

    const v8f vzero = {0.f, 0.f, 0.f, 0.f, 0.f, 0.f, 0.f, 0.f};
    v8f acc[4];
    #pragma unroll
    for (int j = 0; j < 4; ++j) acc[j] = vzero;

    const __bf16* __restrict__ hrow = hB + (size_t)a_row * HID;
    for (int kc = 0; kc < KC_H; ++kc) {
        AFrag a;
        const __bf16* p = hrow + kc * 32 + hi * 16;
        a.h[0] = *(const v8bf*)(p);
        a.h[1] = *(const v8bf*)(p + 8);
        v16bf b[4];
        #pragma unroll
        for (int j = 0; j < 4; ++j) {
            const size_t nt = (size_t)(colBlk * 4 + j);
            b[j] = *(const v16bf*)(wfc + ((size_t)kc * NT_O + nt) * FRAG_ELEMS + lane * 16);
        }
        #pragma unroll
        for (int j = 0; j < 4; ++j) {
            acc[j] = __builtin_amdgcn_wmma_f32_16x16x32_bf16(false, a.v, false, b[j], (short)0, acc[j], false, false);
        }
    }

    #pragma unroll
    for (int j = 0; j < 4; ++j) {
        const int col = colBase + j * 16 + (lane & 15);
        const float bias = fcb[col];
        #pragma unroll
        for (int v = 0; v < 8; ++v) {
            const int row = row0 + hi * 8 + v;
            out[(size_t)row * OUTD + col] = acc[j][v] + bias;
        }
    }
}

// ---------------- launch ----------------
extern "C" void kernel_launch(void* const* d_in, const int* in_sizes, int n_in,
                              void* d_out, int out_size, void* d_ws, size_t ws_size,
                              hipStream_t stream) {
    const float* x   = (const float*)d_in[0];
    const float* xzw = (const float*)d_in[1];
    const float* xzb = (const float*)d_in[2];
    const float* hzw = (const float*)d_in[3];
    const float* hzb = (const float*)d_in[4];
    const float* xrw = (const float*)d_in[5];
    const float* xrb = (const float*)d_in[6];
    const float* hrw = (const float*)d_in[7];
    const float* hrb = (const float*)d_in[8];
    const float* xhw = (const float*)d_in[9];
    const float* xhb = (const float*)d_in[10];
    const float* hhw = (const float*)d_in[11];
    const float* hhb = (const float*)d_in[12];
    const float* fcw = (const float*)d_in[13];
    const float* fcb = (const float*)d_in[14];
    float* out = (float*)d_out;

    // workspace carve-up (256B aligned chunks), total ~147 MB
    char* ws = (char*)d_ws;
    size_t off = 0;
    auto carve = [&](size_t bytes) -> char* {
        char* p = ws + off;
        off += (bytes + 255) & ~(size_t)255;
        return p;
    };
    unsigned* barrier = (unsigned*)carve(256);
    float*    hF      = (float*)   carve((size_t)BATCH * HID * sizeof(float));
    __bf16*   hB0     = (__bf16*)  carve((size_t)BATCH * HID * sizeof(__bf16));
    __bf16*   hB1     = (__bf16*)  carve((size_t)BATCH * HID * sizeof(__bf16));
    __bf16*   xbf     = (__bf16*)  carve((size_t)BATCH * SEQ * INDIM * sizeof(__bf16));
    __bf16*   wp      = (__bf16*)  carve((size_t)KC_TOT * 3 * NT_H * FRAG_ELEMS * sizeof(__bf16));
    __bf16*   wfc     = (__bf16*)  carve((size_t)KC_H * NT_O * FRAG_ELEMS * sizeof(__bf16));
    (void)in_sizes; (void)n_in; (void)out_size; (void)ws_size;

    // 1) init h state + barrier
    {
        int n = BATCH * HID;
        gru_init_kernel<<<(n + 255) / 256, 256, 0, stream>>>(hF, hB0, hB1, barrier);
    }
    // 2) convert x to bf16
    cvt_x_kernel<<<4096, 256, 0, stream>>>(x, xbf);
    // 3) pack gate weights into WMMA B-fragment layout
    {
        size_t n = (size_t)KC_TOT * 3 * NT_H * FRAG_ELEMS;
        pack_gate_weights_kernel<<<(unsigned)((n + 255) / 256), 256, 0, stream>>>(
            xzw, xrw, xhw, hzw, hrw, hhw, wp);
    }
    // 4) pack fc weights
    {
        size_t n = (size_t)KC_H * NT_O * FRAG_ELEMS;
        pack_fc_weights_kernel<<<(unsigned)((n + 255) / 256), 256, 0, stream>>>(fcw, wfc);
    }
    // 5) persistent GRU recurrence over all 512 timesteps (288 KB dynamic LDS)
    gru_persistent_kernel<<<GRU_BLOCKS, 128, LDS_BYTES, stream>>>(
        xbf, wp, xzb, hzb, xrb, hrb, xhb, hhb, hF, hB0, hB1, barrier);
    // 6) final FC (h final lives in hB0 after an even number of swaps)
    fc_kernel<<<FC_BLOCKS, 128, 0, stream>>>(hB0, wfc, fcb, out);
}
